// LaneAttention_27496380629535
// MI455X (gfx1250) — compile-verified, matrix operand order checked by reading, not run on hardware
//
#include <hip/hip_runtime.h>

// CDNA5 (gfx1250) fused lane-attention.
// - All six GEMMs on v_wmma_f32_16x16x32_bf16 (fp32 accumulate).
// - fp32 softmax + fp32 residual.
// - bf16 shadow buffers so GEMM inner loops are pure ds_load_b128 + wmma.
// - Bulk global->LDS staging on the CDNA5 async path:
//   global_load_async_to_lds_b128 + s_wait_asynccnt (ASYNCcnt), with a
//   TDM tensor_load_to_lds variant for the 5-arg ROCm-7.2 toolchain.

typedef __attribute__((ext_vector_type(16))) __bf16 v16bf;
typedef __attribute__((ext_vector_type(8)))  __bf16 v8bf;
typedef __attribute__((ext_vector_type(8)))  float  v8f;
typedef __attribute__((ext_vector_type(4)))  unsigned int v4u;
typedef __attribute__((ext_vector_type(8)))  int v8i;
typedef __attribute__((ext_vector_type(4)))  int v4i;

// ---- CDNA5 staging-path selection (compile-gated, layered) ----
#if defined(__HIP_DEVICE_COMPILE__) && defined(__gfx1250__)
#if __has_builtin(__builtin_amdgcn_tensor_load_to_lds) && \
    __has_builtin(__builtin_amdgcn_s_wait_tensorcnt) && \
    !__has_include(<hip/amd_detail/amd_gfx1250_TDM.h>)
#define USE_TDM 1
#endif
#endif
#ifndef USE_TDM
#define USE_TDM 0
#endif

#if defined(__HIP_DEVICE_COMPILE__) && defined(__gfx1250__)
#if !USE_TDM && __has_builtin(__builtin_amdgcn_global_load_async_to_lds_b128) && \
    __has_builtin(__builtin_amdgcn_s_wait_asynccnt)
#define USE_ASYNC 1
#endif
#endif
#ifndef USE_ASYNC
#define USE_ASYNC 0
#endif

#if defined(__HIP_DEVICE_COMPILE__) && defined(__gfx1250__)
#if USE_TDM
#warning "CDNA5 staging path: TDM tensor_load_to_lds (5-arg)"
#elif USE_ASYNC
#warning "CDNA5 staging path: global_load_async_to_lds_b128 (ASYNCcnt)"
#else
#warning "CDNA5 staging path: plain loads (no TDM/async builtins available)"
#endif
#endif

#if USE_ASYNC
// Builtin prototype (from the round-3 diagnostic): first param is AS(1) v4i*,
// second is AS(3) v4i*, then imm offset + imm cpol.
typedef __attribute__((address_space(1))) v4i gbl_v4i;
typedef __attribute__((address_space(3))) v4i lds_v4i;
// 16B per lane; 32 lanes = one 512B chunk per instruction.
#define ASYNC_CP16(gptr, lptr)                                        \
  __builtin_amdgcn_global_load_async_to_lds_b128(                     \
      (gbl_v4i*)(uintptr_t)(gptr), (lds_v4i*)(lptr), 0, 0)
#endif

namespace {
constexpr int TT = 30;    // len_pred
constexpr int BB = 64;    // batch
constexpr int VV = 32;    // n_vehicles
constexpr int FF = 128;   // feature size
constexpr int LL = 64;    // n_lanes
constexpr int LFD = 64;   // lane feature size
constexpr int HH = 4;     // heads
constexpr int HD = 32;    // head dim
constexpr int THN = 20;   // len_hist

constexpr int PAIRS = 8;                    // (b,v) pairs per block (weight reuse)
constexpr int NTHR = 256;                   // 8 wave32 per workgroup
constexpr int NBLK = (BB * VV) / PAIRS;     // 256 blocks

// LDS pitches (elements): rows 16B-aligned, K-groups of 8 are 16B-aligned.
constexpr int P72 = 72;    // 64 + 8 pad
constexpr int P136 = 136;  // 128 + 8 pad
constexpr int P264 = 264;  // 4*64 + 8 pad

// LDS byte offsets (all multiples of 16)
constexpr int OFF_WK  = 0;                         // Wk bf16 [128][72]
constexpr int OFF_WV  = OFF_WK  + FF * P72 * 2;    // Wv bf16 [128][72]
constexpr int OFF_WQ  = OFF_WV  + FF * P72 * 2;    // Wq bf16 [128][136]
constexpr int OFF_WC  = OFF_WQ  + FF * P136 * 2;   // Wc bf16 [128][136]
constexpr int OFF_LN  = OFF_WC  + FF * P136 * 2;   // lanes bf16 [64][72]
constexpr int OFF_VE  = OFF_LN  + LL * P72 * 2;    // vehicles f32 [32][136] (DMA dest)
constexpr int OFF_VEB = OFF_VE  + 32 * P136 * 4;   // vehicles bf16 [32][136]
constexpr int OFF_Q   = OFF_VEB + 32 * P136 * 2;   // q bf16 [32][136]
constexpr int OFF_K   = OFF_Q   + 32 * P136 * 2;   // k bf16 [64][136]
constexpr int OFF_VT  = OFF_K   + LL * P136 * 2;   // v^T bf16 [128][72]
constexpr int OFF_S   = OFF_VT  + FF * P72 * 2;    // scores f32 [32][264]; also lane-stage f32
constexpr int OFF_PB  = OFF_S   + 32 * P264 * 4;   // probs bf16 [32][264]
constexpr int OFF_OB  = OFF_PB  + 32 * P264 * 2;   // attn-out bf16 [32][136]
constexpr int OFF_M   = OFF_OB  + 32 * P136 * 2;   // mask f32 [64]
constexpr int SMEM_BYTES = OFF_M + LL * 4;         // 246,016 B (< 320KB WGP LDS)
} // namespace

__device__ __forceinline__ __bf16 f2bf(float x) { return (__bf16)x; }

__device__ __forceinline__ v8f zero8() {
  v8f z;
#pragma unroll
  for (int i = 0; i < 8; ++i) z[i] = 0.0f;
  return z;
}

// 16(row/col) x 32(K) bf16 fragment from row-major bf16 LDS.
// ISA layout: lane holds row (lane&15); K elems i<8 -> k0+(lane&16?8:0)+i,
// elems 8..15 -> same +16. Each half is one contiguous 16B v8bf load.
__device__ __forceinline__ v16bf frag_bf16(const __bf16* base, int pitch, int row0, int k0, int lane) {
  const __bf16* p = base + (row0 + (lane & 15)) * pitch + k0 + ((lane & 16) ? 8 : 0);
  v8bf lo = *(const v8bf*)(p);
  v8bf hi = *(const v8bf*)(p + 16);
  v16bf r;
#pragma unroll
  for (int i = 0; i < 8; ++i) { r[i] = lo[i]; r[i + 8] = hi[i]; }
  return r;
}

// D-fragment stores. C/D layout: VGPR j holds (M = m0 + j + (lane&16?8:0), N = n0 + (lane&15)).
__device__ __forceinline__ void st_f32(float* base, int pitch, int m0, int n0, v8f d, int lane) {
  float* p = base + (m0 + ((lane & 16) ? 8 : 0)) * pitch + n0 + (lane & 15);
#pragma unroll
  for (int j = 0; j < 8; ++j) p[j * pitch] = d[j];
}

__device__ __forceinline__ void st_bf16(__bf16* base, int pitch, int m0, int n0, v8f d, int lane) {
  __bf16* p = base + (m0 + ((lane & 16) ? 8 : 0)) * pitch + n0 + (lane & 15);
#pragma unroll
  for (int j = 0; j < 8; ++j) p[j * pitch] = f2bf(d[j]);
}

// Transposed bf16 store: dst[n][m]; one contiguous 16B v8bf store per lane.
__device__ __forceinline__ void st_bf16_T(__bf16* base, int pitch, int m0, int n0, v8f d, int lane) {
  __bf16* p = base + (n0 + (lane & 15)) * pitch + m0 + ((lane & 16) ? 8 : 0);
  v8bf t;
#pragma unroll
  for (int j = 0; j < 8; ++j) t[j] = f2bf(d[j]);
  *(v8bf*)p = t;
}

#define WMMA_BF16(a, b, c) \
  __builtin_amdgcn_wmma_f32_16x16x32_bf16(false, (a), false, (b), (short)0, (c), false, false)

#if USE_TDM
// TDM descriptor for the vehicles tile: 2D tile 30 rows x 128 fp32, row stride
// B*V*F elements in memory, LDS-padded 8 dwords after every 128-dword row so
// it lands directly in the [32][136] fp32 buffer.
__device__ __forceinline__ void tdm_load_vehicles(const float* gsrc, unsigned ldsByteOff) {
  const unsigned long long ga = (unsigned long long)(uintptr_t)gsrc;
  const v4u g0 = {
    1u,                                               // count=1, user descriptor
    ldsByteOff,                                       // lds_addr (bytes)
    (unsigned)ga,                                     // global_addr[31:0]
    (unsigned)((ga >> 32) & 0x01ffffffu) | (2u << 30) // global_addr[56:32] | type=2
  };
  const v8i g1 = {
    (int)((2u << 16)      // data_size = 4B
        | (1u << 20)      // pad_enable
        | (6u << 22)      // pad_interval: 128 dwords (code 6 = 2*2^6)
        | (7u << 25)),    // pad_amount: 8 dwords (code 7 = 8)
    (int)((unsigned)FF << 16),  // tensor_dim0 = 128 -> bits[63:48]
    (int)((unsigned)TT << 16),  // tensor_dim1 = 30  -> low16 at bits[95:80]
    (int)((unsigned)FF << 16),  // tile_dim0 = 128 -> bits[127:112]
    (int)TT,                    // tile_dim1 = 30  -> bits[143:128]
    (int)(BB * VV * FF),        // tensor_dim0_stride = 262144 -> bits[191:160]
    0, 0
  };
  const v4i gz = { 0, 0, 0, 0 };
  __builtin_amdgcn_tensor_load_to_lds(g0, g1, gz, gz, 0);
}
#endif

__global__ __launch_bounds__(NTHR) void lane_attn_wmma(
    const float* __restrict__ veh,     // [T,B,V,F]
    const float* __restrict__ lanes,   // [B,V,L,LF]
    const unsigned char* __restrict__ maskL,  // [TH,B,V,L] bool (1 byte)
    const float* __restrict__ Wk, const float* __restrict__ Wv,  // [F,LF]
    const float* __restrict__ Wq, const float* __restrict__ Wc,  // [F,F]
    float* __restrict__ out)           // [T,B,V,F]
{
  extern __shared__ __align__(16) char smem[];
  __bf16* wKs  = (__bf16*)(smem + OFF_WK);
  __bf16* wVs  = (__bf16*)(smem + OFF_WV);
  __bf16* wQs  = (__bf16*)(smem + OFF_WQ);
  __bf16* wCs  = (__bf16*)(smem + OFF_WC);
  __bf16* sLn  = (__bf16*)(smem + OFF_LN);
  float*  sVe  = (float*)(smem + OFF_VE);
  __bf16* sVeB = (__bf16*)(smem + OFF_VEB);
  __bf16* sQs  = (__bf16*)(smem + OFF_Q);
  __bf16* sKs  = (__bf16*)(smem + OFF_K);
  __bf16* sVt  = (__bf16*)(smem + OFF_VT);
  float*  sS   = (float*)(smem + OFF_S);
  float*  sFin = (float*)(smem + OFF_S);   // alias: scores dead after softmax
  float*  sStage = (float*)(smem + OFF_S); // alias: f32 lane staging before scores exist
  __bf16* sPb  = (__bf16*)(smem + OFF_PB);
  __bf16* sOb  = (__bf16*)(smem + OFF_OB);
  float*  sM   = (float*)(smem + OFF_M);

  const int tid = threadIdx.x;
  const int lane = tid & 31;
  // Scalarize the wave index so all tile loops / branches around WMMA are
  // s_cbranch scalar control flow (EXEC stays all-ones through WMMA blocks).
  const int wave = __builtin_amdgcn_readfirstlane((int)(threadIdx.x >> 5));

  // ---- Stage weights into LDS as bf16 (once per block; shared by 8 pairs) ----
  for (int i = tid; i < FF * LFD; i += NTHR) {
    const int r = i >> 6, c = i & 63;
    wKs[r * P72 + c] = f2bf(Wk[i]);
    wVs[r * P72 + c] = f2bf(Wv[i]);
  }
  for (int i = tid; i < FF * FF; i += NTHR) {
    const int r = i >> 7, c = i & 127;
    wQs[r * P136 + c] = f2bf(Wq[i]);
    wCs[r * P136 + c] = f2bf(Wc[i]);
  }

  for (int pp = 0; pp < PAIRS; ++pp) {
    const int pairId = blockIdx.x * PAIRS + pp;
    const int b = pairId >> 5;   // / VV
    const int v = pairId & 31;   // % VV

    __syncthreads();  // weights ready (pp==0) / previous pair fully consumed

    const float* lp = lanes + (size_t)(b * VV + v) * LL * LFD;

#if USE_ASYNC
    // lanes[b,v] (16KB contiguous) -> f32 staging via async-to-LDS, 512B/op.
    for (int i = wave; i < 32; i += 8)
      ASYNC_CP16(lp + i * 128 + lane * 4, sStage + i * 128 + lane * 4);
    // vehicles rows: one async b128 per row (32 lanes x 16B = 512B row).
    for (int t = wave; t < TT; t += 8)
      ASYNC_CP16(veh + (((size_t)t * BB + b) * VV + v) * FF + lane * 4,
                 sVe + t * P136 + lane * 4);
#else
    // lanes[b,v] -> bf16 LDS directly
    for (int i = tid; i < LL * LFD; i += NTHR)
      sLn[(i >> 6) * P72 + (i & 63)] = f2bf(lp[i]);
#if USE_TDM
    if (wave == 0)
      tdm_load_vehicles(veh + ((size_t)b * VV + v) * FF, (unsigned)OFF_VE);
#else
    for (int i = tid; i < 30 * FF; i += NTHR) {
      const int t = i >> 7, f = i & 127;
      sVe[t * P136 + f] = veh[(((size_t)t * BB + b) * VV + v) * FF + f];
    }
#endif
#endif
    // zero-pad vehicle rows 30,31 (all paths; disjoint from DMA/loads above)
    for (int i = tid; i < 2 * FF; i += NTHR)
      sVe[(TT + (i >> 7)) * P136 + (i & 127)] = 0.0f;

    // mask[l] = any over history
    if (tid < LL) {
      const unsigned char* mp = maskL + (size_t)(b * VV + v) * LL + tid;
      int any = 0;
      for (int th = 0; th < THN; ++th) any |= mp[(size_t)th * BB * VV * LL];
      sM[tid] = any ? 1.0f : 0.0f;
    }

    // prefetch next pair's lane tile (global_prefetch_b8)
    if (pp + 1 < PAIRS)
      __builtin_prefetch(lanes + (size_t)(pairId + 1) * LL * LFD + tid * 16, 0, 3);

#if USE_TDM
    if (wave == 0) __builtin_amdgcn_s_wait_tensorcnt((short)0);  // s_wait_tensorcnt 0
#elif USE_ASYNC
    __builtin_amdgcn_s_wait_asynccnt((short)0);                  // s_wait_asynccnt 0
#endif
    __syncthreads();

#if USE_ASYNC
    // Convert staged f32 lanes -> bf16 operand layout.
    for (int i = tid; i < LL * LFD; i += NTHR)
      sLn[(i >> 6) * P72 + (i & 63)] = f2bf(sStage[i]);
#endif
    // One-time bf16 shadow of vehicles (feeds the Q-proj A operand).
    for (int i = tid; i < 32 * FF; i += NTHR) {
      const int t = i >> 7, f = i & 127;
      sVeB[t * P136 + f] = f2bf(sVe[t * P136 + f]);
    }
    __syncthreads();

    // ---- Stage 1: K = lanes*Wk^T, V^T = (lanes*Wv^T)^T, Q = veh*Wq^T (80 tiles / 8 waves) ----
    for (int ti = wave; ti < 80; ti += 8) {
      if (ti < 64) {
        const int which = ti >> 5, t2 = ti & 31;
        const int m0 = (t2 >> 3) << 4;   // L tile
        const int n0 = (t2 & 7) << 4;    // F tile
        const __bf16* W = which ? wVs : wKs;
        v8f acc = zero8();
#pragma unroll
        for (int k0 = 0; k0 < LFD; k0 += 32) {
          v16bf a  = frag_bf16(sLn, P72, m0, k0, lane);
          v16bf bm = frag_bf16(W,   P72, n0, k0, lane);  // B col n = W row n
          acc = WMMA_BF16(a, bm, acc);
        }
        if (which == 0) st_bf16(sKs, P136, m0, n0, acc, lane);
        else            st_bf16_T(sVt, P72, m0, n0, acc, lane);
      } else {
        const int t2 = ti - 64;
        const int m0 = (t2 >> 3) << 4;   // T tile
        const int n0 = (t2 & 7) << 4;    // F tile
        v8f acc = zero8();
#pragma unroll
        for (int k0 = 0; k0 < FF; k0 += 32) {
          v16bf a  = frag_bf16(sVeB, P136, m0, k0, lane);
          v16bf bm = frag_bf16(wQs, P136, n0, k0, lane);
          acc = WMMA_BF16(a, bm, acc);
        }
        st_bf16(sQs, P136, m0, n0, acc, lane);
      }
    }
    __syncthreads();

    // ---- Stage 2: scores[t,h,l] = q.k/sqrt(hd), masked -> sS[t][h*64+l] (32 tiles) ----
    for (int ti = wave; ti < 32; ti += 8) {
      const int h = ti >> 3, t2 = ti & 7;
      const int m0 = (t2 >> 2) << 4;   // T tile
      const int n0 = (t2 & 3) << 4;    // L tile
      v16bf a  = frag_bf16(sQs, P136, m0, h * HD, lane);
      v16bf bm = frag_bf16(sKs, P136, n0, h * HD, lane);
      v8f acc = zero8();
      acc = WMMA_BF16(a, bm, acc);
      const int nl = n0 + (lane & 15);
      const float keep = sM[nl];
      const float scale = 0.17677669529663687f;  // 1/sqrt(32)
      float* p = sS + (m0 + ((lane & 16) ? 8 : 0)) * P264 + h * LL + nl;
#pragma unroll
      for (int j = 0; j < 8; ++j) p[j * P264] = (keep != 0.0f) ? acc[j] * scale : -1e9f;
    }
    __syncthreads();

    // ---- Stage 3: fp32 softmax over l (128 rows, one thread each); bf16 probs out ----
    if (tid < 32 * HH) {
      float*  row  = sS  + (tid >> 2) * P264 + (tid & 3) * LL;
      __bf16* rowb = sPb + (tid >> 2) * P264 + (tid & 3) * LL;
      float mx = row[0];
      for (int l = 1; l < LL; ++l) mx = fmaxf(mx, row[l]);
      float sum = 0.0f;
      for (int l = 0; l < LL; ++l) { const float e = __expf(row[l] - mx); row[l] = e; sum += e; }
      const float inv = 1.0f / sum;
      for (int l = 0; l < LL; ++l) rowb[l] = f2bf(row[l] * inv);
    }
    __syncthreads();

    // ---- Stage 4: attnOut = P * V (16 tiles; B from v^T so K=L is contiguous) ----
    for (int ti = wave; ti < 16; ti += 8) {
      const int h = ti >> 2, t2 = ti & 3;
      const int m0 = (t2 >> 1) << 4;   // T tile
      const int n0 = (t2 & 1) << 4;    // hd tile
      v8f acc = zero8();
#pragma unroll
      for (int k0 = 0; k0 < LL; k0 += 32) {
        v16bf a  = frag_bf16(sPb, P264, m0, h * LL + k0, lane);
        v16bf bm = frag_bf16(sVt, P72, h * HD + n0, k0, lane);
        acc = WMMA_BF16(a, bm, acc);
      }
      st_bf16(sOb, P136, m0, h * HD + n0, acc, lane);
    }
    __syncthreads();

    // ---- Stage 5: final = attnOut * Wc^T -> sFin (reuses scores LDS) ----
    for (int ti = wave; ti < 16; ti += 8) {
      const int m0 = (ti >> 3) << 4;
      const int n0 = (ti & 7) << 4;
      v8f acc = zero8();
#pragma unroll
      for (int k0 = 0; k0 < FF; k0 += 32) {
        v16bf a  = frag_bf16(sOb, P136, m0, k0, lane);
        v16bf bm = frag_bf16(wCs, P136, n0, k0, lane);
        acc = WMMA_BF16(a, bm, acc);
      }
      st_f32(sFin, P136, m0, n0, acc, lane);
    }
    __syncthreads();

    // ---- Stage 6: residual add + coalesced 512B-row writeout ----
    for (int i = tid; i < TT * FF; i += NTHR) {
      const int t = i >> 7, f = i & 127;
      out[(((size_t)t * BB + b) * VV + v) * FF + f] = sFin[t * P136 + f] + sVe[t * P136 + f];
    }
  }
}

extern "C" void kernel_launch(void* const* d_in, const int* in_sizes, int n_in,
                              void* d_out, int out_size, void* d_ws, size_t ws_size,
                              hipStream_t stream) {
  (void)in_sizes; (void)n_in; (void)out_size; (void)d_ws; (void)ws_size;
  const float* vehicles = (const float*)d_in[0];
  const float* lanes = (const float*)d_in[1];
  const unsigned char* maskL = (const unsigned char*)d_in[2];  // jax bool -> 1 byte/elem
  const float* Wk = (const float*)d_in[3];
  const float* Wv = (const float*)d_in[4];
  const float* Wq = (const float*)d_in[5];
  const float* Wc = (const float*)d_in[6];
  float* out = (float*)d_out;

  lane_attn_wmma<<<dim3(NBLK), dim3(NTHR), SMEM_BYTES, stream>>>(
      vehicles, lanes, maskL, Wk, Wv, Wq, Wc, out);
}
